// Transformer_2413771621095
// MI455X (gfx1250) — matrix operand order repaired and checked by reading, not compiled, not run on hardware
//
#include <hip/hip_runtime.h>
#include <hip/hip_bf16.h>
#include <math.h>

// ---------------- CDNA5 WMMA types ----------------
typedef __attribute__((ext_vector_type(16))) _Float16 v16h;
typedef __attribute__((ext_vector_type(8)))  _Float16 v8h;
typedef __attribute__((ext_vector_type(8)))  float    v8f;
typedef __attribute__((ext_vector_type(4)))  float    vf4;

#define Dm   512
#define Hh   8
#define DHd  64
#define Lseq 1024
#define Bb   2
#define Mrows (Bb * Lseq)          // 2048
#define SCALE 0.125f               // 64^-0.5
#define NEGV  (-1e20f)

__device__ __forceinline__ v8f wmma_f16(v16h a, v16h b, v8f c) {
  return __builtin_amdgcn_wmma_f32_16x16x32_f16(false, a, false, b, (short)0, c,
                                                false, false);
}

// A fragment: 16x32 f16 (M x K), loaded from row-major fp32 matrix.
// This code's contiguous k order: lane m=lane&15, kh=lane>>4; elem e -> K=kh*16+e
__device__ __forceinline__ v16h load_a_frag(const float* __restrict__ A, int lda,
                                            int row0, int k0) {
  const int lane = threadIdx.x & 31;
  const int m = lane & 15, kh = lane >> 4;
  const float* p = A + (size_t)(row0 + m) * lda + k0 + kh * 16;
  v16h r;
#pragma unroll
  for (int g = 0; g < 2; ++g) {
    const float* q = p + g * 8;
    vf4 u0 = *(const vf4*)(q);
    vf4 u1 = *(const vf4*)(q + 4);
#pragma unroll
    for (int t = 0; t < 4; ++t) {
      r[g * 8 + t]     = (_Float16)u0[t];
      r[g * 8 + 4 + t] = (_Float16)u1[t];
    }
  }
  return r;
}

// B fragment: 32x16 f16 (K x N) from row-major fp32 W[K][ldw] at (k0, col0).
// Same contiguous k order as A: lane n=lane&15, kh=lane>>4; elem e -> K=kh*16+e, column n.
// 32-bit inner stride math to avoid v_mul_u64 in the hot loop.
__device__ __forceinline__ v16h load_b_frag_strided(const float* __restrict__ W,
                                                    int ldw, int k0, int col0) {
  const int lane = threadIdx.x & 31;
  const int n = lane & 15, kh = lane >> 4;
  const float* p = W + (size_t)(k0 + kh * 16) * ldw + col0 + n;
  v16h r;
#pragma unroll
  for (int e = 0; e < 16; ++e) r[e] = (_Float16)p[e * ldw];
  return r;
}

// B fragment where B[k][n] lives at base + n*rowstride + k  (i.e. K^T from a
// row-major K matrix) -> fully contiguous 16-float runs per lane.
__device__ __forceinline__ v16h load_b_frag_rows(const float* __restrict__ base,
                                                 int rowstride, int d0) {
  const int lane = threadIdx.x & 31;
  const int n = lane & 15, kh = lane >> 4;
  const float* p = base + (size_t)n * rowstride + d0 + kh * 16;
  v16h r;
#pragma unroll
  for (int q = 0; q < 4; ++q) {
    vf4 u = *(const vf4*)(p + q * 4);
    r[q * 4 + 0] = (_Float16)u[0];
    r[q * 4 + 1] = (_Float16)u[1];
    r[q * 4 + 2] = (_Float16)u[2];
    r[q * 4 + 3] = (_Float16)u[3];
  }
  return r;
}

// ---------------- Kernels ----------------

// out[row][c] = emb[tok[row]][c] + pos[row % 1024][c]
__global__ void embed_kernel(const int* __restrict__ tok,
                             const float* __restrict__ emb,
                             const float* __restrict__ pos,
                             float* __restrict__ out) {
  const int row = blockIdx.x;
  const int l = row & (Lseq - 1);
  const int t = tok[row];
#pragma unroll
  for (int i = 0; i < 2; ++i) {
    const int c = threadIdx.x + i * 256;
    out[(size_t)row * Dm + c] = emb[(size_t)t * Dm + c] + pos[(size_t)l * Dm + c];
  }
}

// One wave per row (8 waves / 256-thread block). O = LN(X [+ R]) * g + b
__global__ void ln_kernel(const float* __restrict__ X, const float* __restrict__ R,
                          const float* __restrict__ g, const float* __restrict__ bb,
                          float* __restrict__ O, int nrows) {
  const int row = blockIdx.x * 8 + (threadIdx.x >> 5);
  if (row >= nrows) return;
  const int lane = threadIdx.x & 31;
  const float* x = X + (size_t)row * Dm;
  const float* rr = R ? R + (size_t)row * Dm : nullptr;
  float v[16];
  float s = 0.f;
#pragma unroll
  for (int i = 0; i < 16; ++i) {
    float t = x[lane + 32 * i];
    if (rr) t += rr[lane + 32 * i];
    v[i] = t;
    s += t;
  }
#pragma unroll
  for (int off = 16; off; off >>= 1) s += __shfl_xor(s, off, 32);
  const float mean = s * (1.0f / Dm);
  float var = 0.f;
#pragma unroll
  for (int i = 0; i < 16; ++i) {
    const float d = v[i] - mean;
    var += d * d;
  }
#pragma unroll
  for (int off = 16; off; off >>= 1) var += __shfl_xor(var, off, 32);
  const float inv = rsqrtf(var * (1.0f / Dm) + 1e-5f);
#pragma unroll
  for (int i = 0; i < 16; ++i) {
    const int c = lane + 32 * i;
    O[(size_t)row * Dm + c] = (v[i] - mean) * inv * g[c] + bb[c];
  }
}

// C[M][N] = act(A[M][K] @ W[K][N] + bias[N]).
// 4-wave block computes a 128x64 C tile (each wave 32 rows -> 2 A-frags x
// 4 B-frags = 8 WMMAs per 32-K panel). The 32x64 fp32 W panel is loaded with
// coalesced b128 loads (each thread: 16 contiguous floats of one k row),
// converted to f16 and stored transposed (Bt[n][k]) in LDS so each wave's B
// fragment is two contiguous ds_load_b128. Next panel prefetched with
// global_prefetch_b8.
__global__ void gemm_bias(const float* __restrict__ A, const float* __restrict__ W,
                          const float* __restrict__ bias, float* __restrict__ C,
                          int K, int N, int act) {
  __shared__ _Float16 Bt[64 * 32];   // [n_local][k_local] f16, 4 KB
  const int col0 = blockIdx.x * 64;
  const int row0 = blockIdx.y * 128;
  const int wave = threadIdx.x >> 5;
  const int lane = threadIdx.x & 31;
  const int n = lane & 15, kh = lane >> 4;
  const int mrow0 = row0 + wave * 32;
  // panel fill assignment: thread -> (k row, 16-wide n chunk)
  const int fk = threadIdx.x >> 2;        // 0..31
  const int fn = (threadIdx.x & 3) * 16;  // 0,16,32,48
  v8f acc[2][4] = {};

  for (int k0 = 0; k0 < K; k0 += 32) {
    __syncthreads();  // previous iteration's Bt reads complete
    {
      const float* Wp = W + (size_t)(k0 + fk) * N + col0 + fn;  // hoisted 64-bit math
      vf4 u[4];
#pragma unroll
      for (int q = 0; q < 4; ++q) u[q] = *(const vf4*)(Wp + q * 4);
      _Float16* bp = &Bt[fn * 32 + fk];   // [n][k], n varies -> stride 32 halves
#pragma unroll
      for (int q = 0; q < 4; ++q)
#pragma unroll
        for (int t = 0; t < 4; ++t) bp[(q * 4 + t) * 32] = (_Float16)u[q][t];
    }
    __syncthreads();
    if (k0 + 32 < K) {  // prefetch next W panel (global_prefetch_b8)
      __builtin_prefetch(W + (size_t)(k0 + 32 + fk) * N + col0 + fn, 0, 1);
    }
    const v16h a0 = load_a_frag(A, K, mrow0, k0);
    const v16h a1 = load_a_frag(A, K, mrow0 + 16, k0);
#pragma unroll
    for (int nt = 0; nt < 4; ++nt) {
      const _Float16* p = &Bt[(nt * 16 + n) * 32 + kh * 16];
      const v8h u0 = *(const v8h*)p;
      const v8h u1 = *(const v8h*)(p + 8);
      v16h bf;
#pragma unroll
      for (int t = 0; t < 8; ++t) { bf[t] = u0[t]; bf[8 + t] = u1[t]; }
      acc[0][nt] = wmma_f16(a0, bf, acc[0][nt]);
      acc[1][nt] = wmma_f16(a1, bf, acc[1][nt]);
    }
  }

#pragma unroll
  for (int mt = 0; mt < 2; ++mt) {
#pragma unroll
    for (int nt = 0; nt < 4; ++nt) {
      const float bv = bias[col0 + nt * 16 + n];
#pragma unroll
      for (int r = 0; r < 8; ++r) {
        float v = acc[mt][nt][r] + bv;
        if (act) v = 0.5f * v * (1.0f + erff(v * 0.70710678118654752f));
        C[(size_t)(mrow0 + mt * 16 + kh * 8 + r) * N + col0 + nt * 16 + n] = v;
      }
    }
  }
}

// One wave per (b, h, 16-row q tile). mode 0: key-padding mask via enc_tok;
// mode 1: causal. Q/K/V are [b*L, 512] with head h at columns h*64..h*64+63.
// Output in the reference's faithful [b, h, Lq, dh]-contiguous layout.
__global__ void attention_kernel(const float* __restrict__ Q,
                                 const float* __restrict__ Km,
                                 const float* __restrict__ Vm,
                                 const int* __restrict__ enc_tok,
                                 float* __restrict__ Out, int mode) {
  extern __shared__ float sm[];  // 16 x 1024 fp32 score/prob rows
  const int q0 = blockIdx.x * 16;
  const int h = blockIdx.y;
  const int b = blockIdx.z;
  const int lane = threadIdx.x & 31;
  const int n = lane & 15, kh = lane >> 4;
  const size_t base_b = (size_t)b * Lseq * Dm;
  const float* Qh = Q + base_b + h * DHd;
  const float* Kh = Km + base_b + h * DHd;
  const float* Vh = Vm + base_b + h * DHd;

  // ---- scores S = Q K^T, mask BEFORE scale (reference quirk) ----
  for (int kt = 0; kt < Lseq / 16; ++kt) {
    v8f acc = {};
#pragma unroll
    for (int d0 = 0; d0 < DHd; d0 += 32) {
      v16h a = load_a_frag(Qh, Dm, q0, d0);
      v16h bf = load_b_frag_rows(Kh + (size_t)(kt * 16) * Dm, Dm, d0);
      acc = wmma_f16(a, bf, acc);
    }
    const int kpos = kt * 16 + n;
    bool keyok = true;
    if (mode == 0) keyok = (enc_tok[b * Lseq + kpos] != 0);
#pragma unroll
    for (int r = 0; r < 8; ++r) {
      const int qrow = kh * 8 + r;
      const bool ok = (mode == 0) ? keyok : (kpos <= q0 + qrow);
      sm[qrow * Lseq + kpos] = (ok ? acc[r] : NEGV) * SCALE;
    }
  }

  // ---- row softmax over Lk = 1024 (whole wave per row) ----
  for (int r = 0; r < 16; ++r) {
    float* row = sm + r * Lseq;
    float mx = -3.0e38f;
    for (int c = lane; c < Lseq; c += 32) mx = fmaxf(mx, row[c]);
#pragma unroll
    for (int off = 16; off; off >>= 1) mx = fmaxf(mx, __shfl_xor(mx, off, 32));
    float sum = 0.f;
    for (int c = lane; c < Lseq; c += 32) {
      const float e = __expf(row[c] - mx);
      row[c] = e;
      sum += e;
    }
#pragma unroll
    for (int off = 16; off; off >>= 1) sum += __shfl_xor(sum, off, 32);
    const float inv = 1.0f / sum;
    for (int c = lane; c < Lseq; c += 32) row[c] *= inv;
  }

  // ---- O = P V  (16 x 64), faithful [b,h,Lq,dh] output layout ----
  const size_t obase = (((size_t)b * Hh + h) * Lseq + q0) * DHd;
#pragma unroll
  for (int nt = 0; nt < 4; ++nt) {
    v8f acc = {};
    for (int kt = 0; kt < Lseq / 32; ++kt) {
      v16h a = load_a_frag(sm, Lseq, 0, kt * 32);
      v16h bf = load_b_frag_strided(Vh, Dm, kt * 32, nt * 16);
      acc = wmma_f16(a, bf, acc);
    }
#pragma unroll
    for (int r = 0; r < 8; ++r)
      Out[obase + (size_t)(kh * 8 + r) * DHd + nt * 16 + n] = acc[r];
  }
}

// ---------------- Host orchestration ----------------
extern "C" void kernel_launch(void* const* d_in, const int* in_sizes, int n_in,
                              void* d_out, int out_size, void* d_ws, size_t ws_size,
                              hipStream_t stream) {
  (void)in_sizes; (void)n_in; (void)out_size; (void)ws_size;
  auto f = [&](int i) { return (const float*)d_in[i]; };
  const int* enc_tok = (const int*)d_in[0];
  const int* dec_tok = (const int*)d_in[1];
  const float* enc_emb = f(2);
  const float* dec_emb = f(3);
  const float* enc_pos = f(4);
  const float* dec_pos = f(5);
  const float* enc_ng = f(6);
  const float* enc_nb = f(7);
  const float* dec_ng = f(8);
  const float* dec_nb = f(9);

  float* ws = (float*)d_ws;
  const size_t SZ = (size_t)Mrows * Dm;
  float* x    = ws + 0 * SZ;  // encoder activations
  float* y    = ws + 1 * SZ;  // decoder activations
  float* t_ln = ws + 2 * SZ;
  float* t_q  = ws + 3 * SZ;
  float* t_k  = ws + 4 * SZ;
  float* t_v  = ws + 5 * SZ;
  float* t_at = ws + 6 * SZ;
  float* t_o  = ws + 7 * SZ;
  float* t_kn = ws + 8 * SZ;  // cross-attn LN(encoder output)

  const dim3 gGemm(Dm / 64, Mrows / 128);
  const dim3 gAttn(Lseq / 16, Hh, Bb);
  const int LN_GRID = Mrows / 8;

  auto LN = [&](const float* X, const float* R, const float* g, const float* b,
                float* O) {
    ln_kernel<<<LN_GRID, 256, 0, stream>>>(X, R, g, b, O, Mrows);
  };
  auto GEMM = [&](const float* A, const float* W, const float* bias, float* C,
                  int act) {
    gemm_bias<<<gGemm, 128, 0, stream>>>(A, W, bias, C, Dm, Dm, act);
  };
  auto ATTN = [&](const float* Qp, const float* Kp, const float* Vp,
                  const int* tok, float* O, int mode) {
    attention_kernel<<<gAttn, 32, 16 * Lseq * sizeof(float), stream>>>(
        Qp, Kp, Vp, tok, O, mode);
  };

  // ---------------- Encoder ----------------
  embed_kernel<<<Mrows, 256, 0, stream>>>(enc_tok, enc_emb, enc_pos, x);
  for (int i = 0; i < 6; ++i) {
    const int p = 10 + i * 16;  // attn: +0..9, ff: +10..15
    LN(x, nullptr, f(p + 0), f(p + 1), t_ln);
    GEMM(t_ln, f(p + 2), f(p + 3), t_q, 0);
    GEMM(t_ln, f(p + 4), f(p + 5), t_k, 0);
    GEMM(t_ln, f(p + 6), f(p + 7), t_v, 0);
    ATTN(t_q, t_k, t_v, enc_tok, t_at, 0);
    GEMM(t_at, f(p + 8), f(p + 9), t_o, 0);
    LN(t_o, x, enc_ng, enc_nb, x);
    LN(x, nullptr, f(p + 10), f(p + 11), t_ln);
    GEMM(t_ln, f(p + 12), f(p + 13), t_q, 1);  // GELU
    GEMM(t_q, f(p + 14), f(p + 15), t_o, 0);
    LN(t_o, x, enc_ng, enc_nb, x);
  }

  // ---------------- Decoder ----------------
  embed_kernel<<<Mrows, 256, 0, stream>>>(dec_tok, dec_emb, dec_pos, y);
  for (int i = 0; i < 6; ++i) {
    const int p = 10 + 96 + i * 16;
    // self-attention (causal), same weights as cross-attention
    LN(y, nullptr, f(p + 0), f(p + 1), t_ln);
    GEMM(t_ln, f(p + 2), f(p + 3), t_q, 0);
    GEMM(t_ln, f(p + 4), f(p + 5), t_k, 0);
    GEMM(t_ln, f(p + 6), f(p + 7), t_v, 0);
    ATTN(t_q, t_k, t_v, dec_tok, t_at, 1);
    GEMM(t_at, f(p + 8), f(p + 9), t_o, 0);
    LN(t_o, y, dec_ng, dec_nb, y);
    // cross-attention: q from y, k/v from encoder output (LN'd with THIS
    // layer's attn LN params, as in the reference)
    LN(y, nullptr, f(p + 0), f(p + 1), t_ln);
    LN(x, nullptr, f(p + 0), f(p + 1), t_kn);
    GEMM(t_ln, f(p + 2), f(p + 3), t_q, 0);
    GEMM(t_kn, f(p + 4), f(p + 5), t_k, 0);
    GEMM(t_kn, f(p + 6), f(p + 7), t_v, 0);
    ATTN(t_q, t_k, t_v, enc_tok, t_at, 0);
    GEMM(t_at, f(p + 8), f(p + 9), t_o, 0);
    LN(t_o, y, dec_ng, dec_nb, y);
    // feed-forward
    LN(y, nullptr, f(p + 10), f(p + 11), t_ln);
    GEMM(t_ln, f(p + 12), f(p + 13), t_q, 1);  // GELU
    GEMM(t_q, f(p + 14), f(p + 15), t_o, 0);
    LN(t_o, y, dec_ng, dec_nb, (i == 5) ? (float*)d_out : y);
  }
}
